// ls_37538014167151
// MI455X (gfx1250) — compile-verified, matrix operand order checked by reading, not run on hardware
//
#include <hip/hip_runtime.h>

#define TQ 1024
#define DD 768
#define BB 32
#define LL 32
#define DH 384                      // DD / 2 (per-block D half for 48KB LDS)
#define CH 64                       // tokens per proto-chunk
#define NCH (TQ / CH)               // 16 chunks
#define INV_TEMP (1.0f / 0.07f)

typedef __attribute__((ext_vector_type(16))) __bf16 v16bf;
typedef __attribute__((ext_vector_type(8)))  float  v8f;

// ---- workspace layout (elements) ----
#define ACC_N   (BB * LL * DD)      // f32 accumulators
#define PROTO_N (BB * LL * DD)      // bf16 prototypes

__global__ void ls_zero_kernel(float* acc, int* counts, int* labmax, float* out) {
  const int idx = blockIdx.x * blockDim.x + threadIdx.x;
  for (int i = idx; i < ACC_N; i += gridDim.x * blockDim.x) acc[i] = 0.0f;
  if (idx < BB * LL) counts[idx] = 0;
  if (idx < BB)      labmax[idx] = 0;
  if (idx == 0)      out[0] = 0.0f;
}

// 1024 blocks: (dialogue, token-chunk of 64, D-half). 128 threads.
// Chunk-local segment sums in LDS (thread owns d == tid mod 128 -> no races),
// committed with global f32 atomics; labels absent from the chunk are skipped.
__global__ void ls_proto_partial_kernel(const float* __restrict__ feat,
                                        const int*   __restrict__ lens,
                                        const int*   __restrict__ labels,
                                        float* __restrict__ acc,
                                        int*   __restrict__ counts,
                                        int*   __restrict__ labmax) {
  __shared__ float sacc[LL * DH];    // 48 KB
  __shared__ int   cnt[LL];
  const int blk   = blockIdx.x;
  const int b     = blk >> 5;
  const int chunk = (blk & 31) >> 1;
  const int dhalf = blk & 1;
  const int tid   = threadIdx.x;

  const int len = lens[b];
  const int t0  = chunk * CH;
  if (t0 >= len) return;
  const int t1  = (t0 + CH < len) ? (t0 + CH) : len;

  for (int l = 0; l < LL; ++l)
    for (int d = tid; d < DH; d += 128)
      sacc[l * DH + d] = 0.0f;
  if (tid < LL) cnt[tid] = 0;
  __syncthreads();

  const float* fb = feat + (size_t)b * TQ * DD + dhalf * DH;
  const int*   lb = labels + b * TQ;
  int lm = 0;
  for (int t = t0; t < t1; ++t) {
    const int l = lb[t];
    const float* fr = fb + (size_t)t * DD;
    float* ar = sacc + l * DH;
    for (int d = tid; d < DH; d += 128)
      ar[d] += fr[d];
    if (tid == 0) { cnt[l]++; lm = (l > lm) ? l : lm; }
  }
  __syncthreads();

  for (int l = 0; l < LL; ++l) {
    if (cnt[l] == 0) continue;       // label absent from this chunk
    float* gr = acc + ((size_t)(b * LL + l)) * DD + dhalf * DH;
    for (int d = tid; d < DH; d += 128)
      atomicAdd(&gr[d], sacc[l * DH + d]);
  }
  if (dhalf == 0 && tid < LL && cnt[tid] > 0)
    atomicAdd(&counts[b * LL + tid], cnt[tid]);
  if (dhalf == 0 && tid == 0)
    atomicMax(&labmax[b], lm);
}

// divide by counts, emit bf16 prototypes
__global__ void ls_proto_final_kernel(const float* __restrict__ acc,
                                      const int*   __restrict__ counts,
                                      __bf16* __restrict__ protos) {
  const int idx = blockIdx.x * blockDim.x + threadIdx.x;
  if (idx >= PROTO_N) return;
  const int bl = idx / DD;           // b * LL + l
  const int c  = counts[bl];
  const float inv = 1.0f / (float)((c > 0) ? c : 1);
  protos[idx] = (__bf16)(acc[idx] * inv);
}

// 256 blocks (B=32 dialogues x 8 token-chunks of 128). 8 waves per block;
// each wave computes a 16(token) x 32(label) logit tile with two
// v_wmma_f32_16x16x32_bf16 per K-step (K = 768 in 24 steps).
__global__ void ls_wmma_loss_kernel(const float* __restrict__ feat,
                                    const int*   __restrict__ lens,
                                    const int*   __restrict__ labels,
                                    const __bf16* __restrict__ protos,
                                    const int*   __restrict__ labmax,
                                    float* __restrict__ out) {
  __shared__ float slog[128 * LL];   // 16 KB logits staging
  const int b      = blockIdx.x >> 3;
  const int tchunk = blockIdx.x & 7;
  const int tid    = threadIdx.x;
  const int wave   = tid >> 5;
  const int lane   = tid & 31;
  const int m      = lane & 15;
  const int khalf  = (lane >> 4) << 3;                // lanes 16-31 start at K=8
  const int token  = tchunk * 128 + wave * 16 + m;

  const float*  frow = feat   + ((size_t)(b * TQ + token)) * DD;
  const __bf16* p0   = protos + ((size_t)(b * LL + m))      * DD;
  const __bf16* p1   = protos + ((size_t)(b * LL + m + 16)) * DD;

  v8f c0 = {}; v8f c1 = {};
  for (int kk = 0; kk < DD; kk += 32) {
    // prefetch next feature K-tile (speculative; OOB-safe)
    __builtin_prefetch(frow + kk + 32, 0, 0);

    // ISA 16-bit A layout (lane group khalf): VGPR v<4 -> K = khalf+2v(,+1);
    // v>=4 -> K = khalf+16+2(v-4)(,+1): four contiguous float4 spans.
    const float4 f0 = *(const float4*)(frow + kk + khalf);
    const float4 f1 = *(const float4*)(frow + kk + khalf + 4);
    const float4 f2 = *(const float4*)(frow + kk + khalf + 16);
    const float4 f3 = *(const float4*)(frow + kk + khalf + 20);

    union { v16bf v; __bf16 h[16]; } A;
    A.h[0]  = (__bf16)f0.x; A.h[1]  = (__bf16)f0.y;
    A.h[2]  = (__bf16)f0.z; A.h[3]  = (__bf16)f0.w;
    A.h[4]  = (__bf16)f1.x; A.h[5]  = (__bf16)f1.y;
    A.h[6]  = (__bf16)f1.z; A.h[7]  = (__bf16)f1.w;
    A.h[8]  = (__bf16)f2.x; A.h[9]  = (__bf16)f2.y;
    A.h[10] = (__bf16)f2.z; A.h[11] = (__bf16)f2.w;
    A.h[12] = (__bf16)f3.x; A.h[13] = (__bf16)f3.y;
    A.h[14] = (__bf16)f3.z; A.h[15] = (__bf16)f3.w;

    // B layout mirrors A with lane -> column: two uint4 (8 bf16 each) spans.
    union { v16bf v; uint4 q[2]; } B0, B1;
    B0.q[0] = *(const uint4*)(p0 + kk + khalf);
    B0.q[1] = *(const uint4*)(p0 + kk + khalf + 16);
    B1.q[0] = *(const uint4*)(p1 + kk + khalf);
    B1.q[1] = *(const uint4*)(p1 + kk + khalf + 16);

    c0 = __builtin_amdgcn_wmma_f32_16x16x32_bf16(false, A.v, false, B0.v,
                                                 (short)0, c0, false, false);
    c1 = __builtin_amdgcn_wmma_f32_16x16x32_bf16(false, A.v, false, B1.v,
                                                 (short)0, c1, false, false);
  }

  // C layout: VGPR r -> M = r (lanes 0-15) / r+8 (lanes 16-31); N = lane & 15
  const int mrow = (lane < 16) ? 0 : 8;
#pragma unroll
  for (int r = 0; r < 8; ++r) {
    slog[(wave * 16 + mrow + r) * LL + m]      = c0[r];
    slog[(wave * 16 + mrow + r) * LL + m + 16] = c1[r];
  }
  __syncthreads();

  if (tid < 128) {
    const int tok  = tchunk * 128 + tid;
    const float* row = slog + tid * LL;
    const int len  = lens[b];
    const int lab  = labels[b * TQ + tok];
    const int ncol = labmax[b] + 1;      // cols beyond lab_max are -inf
    float mx = -3.402823466e38f;
    for (int l = 0; l < ncol; ++l) mx = fmaxf(mx, row[l] * INV_TEMP);
    float s = 0.0f;
    for (int l = 0; l < ncol; ++l) s += __expf(row[l] * INV_TEMP - mx);
    const float pos = row[lab] * INV_TEMP - mx - __logf(s);
    const float contrib =
        (tok < len) ? (-pos) / ((float)len * (float)BB) : 0.0f;
    atomicAdd(out, contrib);
  }
}

extern "C" void kernel_launch(void* const* d_in, const int* in_sizes, int n_in,
                              void* d_out, int out_size, void* d_ws, size_t ws_size,
                              hipStream_t stream) {
  const float* feat   = (const float*)d_in[0];
  const int*   lens   = (const int*)d_in[1];
  const int*   labels = (const int*)d_in[2];
  float*       out    = (float*)d_out;

  char* ws = (char*)d_ws;
  float*  acc    = (float*)ws;                                   // 3 MB
  __bf16* protos = (__bf16*)(ws + (size_t)ACC_N * sizeof(float)); // 1.5 MB
  int*    counts = (int*)(ws + (size_t)ACC_N * sizeof(float)
                             + (size_t)PROTO_N * sizeof(__bf16));
  int*    labmax = counts + BB * LL;

  ls_zero_kernel<<<(ACC_N + 255) / 256, 256, 0, stream>>>(acc, counts, labmax, out);
  ls_proto_partial_kernel<<<BB * NCH * 2, 128, 0, stream>>>(feat, lens, labels,
                                                            acc, counts, labmax);
  ls_proto_final_kernel<<<(PROTO_N + 255) / 256, 256, 0, stream>>>(acc, counts,
                                                                   protos);
  ls_wmma_loss_kernel<<<BB * (TQ / 128), 256, 0, stream>>>(feat, lens, labels,
                                                           protos, labmax, out);
}
